// WindowAttention_82265803587940
// MI455X (gfx1250) — compile-verified, matrix operand order checked by reading, not run on hardware
//
#include <hip/hip_runtime.h>
#include <hip/hip_bf16.h>

// ---------------------------------------------------------------------------
// Swin-style window attention, fully fused, bf16 WMMA (gfx1250 / CDNA5).
//   x[4096,49,128] -> qkv -> per-head attn (+bias,+mask, softmax x2) -> proj -> GELU
// One workgroup (8 wave32) per window. All intermediates in LDS.
// Fragment loads = 2x ds_load_b128 per lane (K-contiguous layouts everywhere).
// ---------------------------------------------------------------------------

typedef __bf16 bf16_t;
typedef __attribute__((ext_vector_type(16))) __bf16 v16bf;
typedef __attribute__((ext_vector_type(8)))  __bf16 v8bf;
typedef __attribute__((ext_vector_type(8)))  float  v8f;
typedef __attribute__((ext_vector_type(4)))  float  v4f;

#define NTOK   49
#define NPAD   64
#define CDIM   128
#define NH     4
#define HD     32
#define THREADS 256

// LDS strides (elements). bf16 strides are multiples of 8 (16B alignment for
// b128 fragment loads); ALD is a multiple of 4 (16B-aligned fp32 rows).
#define XLD    136   // x tile       [64][XLD]    bf16 (A, row-major)
#define QLD    136   // q, k         [64][QLD]    bf16 (A / B-kmaj)
#define VTLD   72    // v transposed [128][VTLD]  bf16 (B, n-major k-contig)
#define OLD    136   // attn output  [64][OLD]    bf16 (A, row-major)
#define WTLD   40    // qkv_w panel  [384][WTLD]  bf16 (B, n-major k-contig)
#define PLD    72    // probs        [4][64][PLD] bf16 (A, row-major)
#define PWTLD  136   // proj_w^T     [128][PWTLD] bf16 (B, n-major k-contig)
#define ALD    68    // scores fp32  [4][64][ALD]

// byte offsets into dynamic LDS (all 16B aligned)
#define OFF_Q     0                               // 64*136*2 = 17408
#define OFF_K     17408                           // 17408
#define OFF_VT    34816                           // 128*72*2 = 18432
#define OFF_OUT   53248                           // 17408
#define OFF_R1    70656                           // union: xbf+wpanT | pbf | pwT
#define OFF_WPAN  (OFF_R1 + 64*XLD*2)             // = 88064, 384*40*2 = 30720
#define OFF_R2    118784                          // scores, 4*64*68*4 = 69632
#define LDS_TOTAL 188416

__device__ __forceinline__ v8f wmma_bf16(v16bf a, v16bf b, v8f c) {
    return __builtin_amdgcn_wmma_f32_16x16x32_bf16(
        false, a, false, b, (short)0, c, false, false);
}

__device__ __forceinline__ v8f zero_v8f() {
    v8f c;
#pragma unroll
    for (int e = 0; e < 8; ++e) c[e] = 0.0f;
    return c;
}

__device__ __forceinline__ unsigned pack_bf16(float a, float b) {
    unsigned short lo = __builtin_bit_cast(unsigned short, (__bf16)a);
    unsigned short hi = __builtin_bit_cast(unsigned short, (__bf16)b);
    return ((unsigned)hi << 16) | (unsigned)lo;
}

// Two 16B LDS loads -> one 16-element fragment, no repacking moves.
__device__ __forceinline__ v16bf ld_frag_pair(const bf16_t* p) {
    v8bf lo = *(const v8bf*)p;
    v8bf hi = *(const v8bf*)(p + 16);
    return __builtin_shufflevector(lo, hi, 0,1,2,3,4,5,6,7,8,9,10,11,12,13,14,15);
}

// A fragment (16x32): A[m][k] = p[m*rs + k], k contiguous, rs % 8 == 0.
// Lane<16: m=lane, K={0..7,16..23}; lane>=16: m=lane-16, K={8..15,24..31}.
__device__ __forceinline__ v16bf load_frag_a_row(const bf16_t* p, int rs) {
    const int lane = threadIdx.x & 31;
    const int m  = lane & 15;
    const int kb = (lane & 16) ? 8 : 0;
    return ld_frag_pair(p + m * rs + kb);
}

// B fragment (32x16): B[k][n] = p[n*ns + k], k contiguous, ns % 8 == 0.
__device__ __forceinline__ v16bf load_frag_b_kmaj(const bf16_t* p, int ns) {
    const int lane = threadIdx.x & 31;
    const int n  = lane & 15;
    const int kb = (lane & 16) ? 8 : 0;
    return ld_frag_pair(p + n * ns + kb);
}

__global__ __launch_bounds__(THREADS)
void swin_window_attn_kernel(const float* __restrict__ x,
                             const float* __restrict__ mask,
                             const float* __restrict__ qkv_w,
                             const float* __restrict__ qkv_b,
                             const float* __restrict__ proj_w,
                             const float* __restrict__ proj_b,
                             const float* __restrict__ bias_table,
                             const int*   __restrict__ rel_idx,
                             float*       __restrict__ out)
{
    extern __shared__ char lds[];
    bf16_t* qbf   = (bf16_t*)(lds + OFF_Q);     // [64][QLD]       row-major
    bf16_t* kbf   = (bf16_t*)(lds + OFF_K);     // [64][QLD]       row-major
    bf16_t* vT    = (bf16_t*)(lds + OFF_VT);    // [128][VTLD]     n-major
    bf16_t* obf   = (bf16_t*)(lds + OFF_OUT);   // [64][OLD]       row-major
    bf16_t* xbf   = (bf16_t*)(lds + OFF_R1);    // [64][XLD]       stage 1
    bf16_t* wpanT = (bf16_t*)(lds + OFF_WPAN);  // [384][WTLD]     stage 1
    bf16_t* pbf   = (bf16_t*)(lds + OFF_R1);    // [4][64][PLD]    stage 2 (reuse R1)
    bf16_t* pwT   = (bf16_t*)(lds + OFF_R1);    // [128][PWTLD]    stage 3 (reuse R1)
    float*  attn  = (float*)(lds + OFF_R2);     // [4][64][ALD]

    const int tid  = threadIdx.x;
    const int wave = tid >> 5;
    const int lane = tid & 31;
    const int fn   = lane & 15;                 // C-fragment column
    const int fmb  = (lane & 16) ? 8 : 0;       // C-fragment row base
    const int b    = blockIdx.x;
    const int widx = b & 63;                    // shifted-window mask index
    const float scale = 0.1767766952966369f;    // 32^-0.5

    // ---- stage x -> LDS bf16 (packed pair stores), zero-pad rows 49..63 -----
    const float* xg = x + (size_t)b * NTOK * CDIM;
    for (int idx = tid; idx < NTOK * CDIM / 2; idx += THREADS) {
        int r = idx >> 6, c2 = (idx & 63) * 2;
        const float* s = xg + r * CDIM + c2;
        *(unsigned*)(xbf + r * XLD + c2) = pack_bf16(s[0], s[1]);
    }
    for (int idx = tid; idx < (NPAD - NTOK) * CDIM / 2; idx += THREADS) {
        int r = NTOK + (idx >> 6), c2 = (idx & 63) * 2;
        *(unsigned*)(xbf + r * XLD + c2) = 0u;
    }
    __syncthreads();

    // ---- GEMM1: qkv = x[64x128] @ qkv_w[128x384]  (96 tiles of 16x16) -------
    // wave w owns tiles t = w + 8*i (i<12): mt = w&3 constant -> A reuse.
    const int mt1 = wave & 3;
    v8f acc[12];
#pragma unroll
    for (int i = 0; i < 12; ++i) acc[i] = zero_v8f();

    for (int kp = 0; kp < 4; ++kp) {            // K panels of 32
        // stage panel transposed: wpanT[col][k] (k contiguous; packed k-pairs)
        for (int idx = tid; idx < 16 * 384; idx += THREADS) {
            int r2 = idx / 384, c = idx - r2 * 384;
            const float* s = qkv_w + (kp * 32 + 2 * r2) * 384 + c;
            *(unsigned*)(wpanT + c * WTLD + 2 * r2) = pack_bf16(s[0], s[384]);
        }
        __syncthreads();
        v16bf a = load_frag_a_row(xbf + mt1 * 16 * XLD + kp * 32, XLD);
#pragma unroll
        for (int i = 0; i < 12; ++i) {
            int nt = (wave + 8 * i) >> 2;       // 0..23
            v16bf bf = load_frag_b_kmaj(wpanT + nt * 16 * WTLD, WTLD);
            acc[i] = wmma_bf16(a, bf, acc[i]);
        }
        __syncthreads();
    }

    // epilogue: + qkv_b, scale q, cast bf16; q,k row-major, v transposed
#pragma unroll
    for (int i = 0; i < 12; ++i) {
        int nt    = (wave + 8 * i) >> 2;
        int col   = nt * 16 + fn;               // 0..383
        int which = col >> 7;                   // 0=q 1=k 2=v
        int c128  = col & 127;
        int m0    = mt1 * 16 + fmb;
        float bia = qkv_b[col];
        if (which == 0) {
#pragma unroll
            for (int r = 0; r < 8; ++r)
                qbf[(m0 + r) * QLD + c128] = (bf16_t)((acc[i][r] + bia) * scale);
        } else if (which == 1) {
#pragma unroll
            for (int r = 0; r < 8; ++r)
                kbf[(m0 + r) * QLD + c128] = (bf16_t)(acc[i][r] + bia);
        } else {                                // v: packed pairs along M
#pragma unroll
            for (int r = 0; r < 8; r += 2)
                *(unsigned*)(vT + c128 * VTLD + m0 + r) =
                    pack_bf16(acc[i][r] + bia, acc[i][r + 1] + bia);
        }
    }
    __syncthreads();

    // ---- GEMM2: scores[h] = q_h[64x32] @ k_h^T[32x64] -----------------------
    {
        int h = wave >> 1;
        v16bf afr[4];
#pragma unroll
        for (int m = 0; m < 4; ++m)
            afr[m] = load_frag_a_row(qbf + m * 16 * QLD + h * HD, QLD);
#pragma unroll
        for (int nn = 0; nn < 2; ++nn) {
            int nt = (wave & 1) * 2 + nn;
            // B[k][n] = k[nt*16+n][h*32+k]: row-major k is k-contiguous per n
            v16bf bf = load_frag_b_kmaj(kbf + nt * 16 * QLD + h * HD, QLD);
#pragma unroll
            for (int m = 0; m < 4; ++m) {
                v8f c = zero_v8f();
                c = wmma_bf16(afr[m], bf, c);
                float* ap = attn + (h * 64 + m * 16) * ALD + nt * 16;
#pragma unroll
                for (int r = 0; r < 8; ++r) ap[(fmb + r) * ALD + fn] = c[r];
            }
        }
    }
    __syncthreads();

    // ---- softmax row in registers: +bias, +mask, softmax, softmax again -----
    {
        int h = tid >> 6, i = tid & 63;
        unsigned* prow32 = (unsigned*)(pbf + (h * 64 + i) * PLD);
        if (i < NTOK) {
            const float* arow = attn + (h * 64 + i) * ALD;   // 16B aligned
            const int*   ridx = rel_idx + i * NTOK;
            const float* mrow = mask + ((size_t)widx * NTOK + i) * NTOK;
            float rv[NTOK];
#pragma unroll
            for (int j4 = 0; j4 < 12; ++j4) {                // wide row read
                v4f t = *(const v4f*)(arow + 4 * j4);
                rv[4 * j4 + 0] = t[0]; rv[4 * j4 + 1] = t[1];
                rv[4 * j4 + 2] = t[2]; rv[4 * j4 + 3] = t[3];
            }
            rv[48] = arow[48];
            float mx = -3.0e38f;
#pragma unroll
            for (int j = 0; j < NTOK; ++j) {
                float s = rv[j] + bias_table[ridx[j] * NH + h] + mrow[j];
                rv[j] = s;
                mx = fmaxf(mx, s);
            }
            float sum = 0.0f;
#pragma unroll
            for (int j = 0; j < NTOK; ++j) { rv[j] = __expf(rv[j] - mx); sum += rv[j]; }
            float inv = 1.0f / sum;
            float mx2 = -3.0e38f;
#pragma unroll
            for (int j = 0; j < NTOK; ++j) { rv[j] *= inv; mx2 = fmaxf(mx2, rv[j]); }
            float s2 = 0.0f;
#pragma unroll
            for (int j = 0; j < NTOK; ++j) { rv[j] = __expf(rv[j] - mx2); s2 += rv[j]; }
            float inv2 = 1.0f / s2;
#pragma unroll
            for (int j2 = 0; j2 < 24; ++j2)
                prow32[j2] = pack_bf16(rv[2 * j2] * inv2, rv[2 * j2 + 1] * inv2);
            prow32[24] = pack_bf16(rv[48] * inv2, 0.0f);
#pragma unroll
            for (int j2 = 25; j2 < 32; ++j2) prow32[j2] = 0u;
        } else {
#pragma unroll
            for (int j2 = 0; j2 < 32; ++j2) prow32[j2] = 0u;
        }
    }
    __syncthreads();

    // ---- GEMM3: out_h = P_h[64x64] @ v_h[64x32] -----------------------------
    {
        int h  = wave >> 1;
        int nt = wave & 1;
        v8f c[4];
#pragma unroll
        for (int m = 0; m < 4; ++m) c[m] = zero_v8f();
#pragma unroll
        for (int ks = 0; ks < 2; ++ks) {
            // B[k][n] = vT[h*32+nt*16+n][ks*32+k]: k contiguous per column
            v16bf bf = load_frag_b_kmaj(vT + (h * HD + nt * 16) * VTLD + ks * 32, VTLD);
#pragma unroll
            for (int m = 0; m < 4; ++m) {
                v16bf a = load_frag_a_row(pbf + (h * 64 + m * 16) * PLD + ks * 32, PLD);
                c[m] = wmma_bf16(a, bf, c[m]);
            }
        }
#pragma unroll
        for (int m = 0; m < 4; ++m) {
            bf16_t* op = obf + m * 16 * OLD + h * HD + nt * 16;
#pragma unroll
            for (int r = 0; r < 8; ++r) op[(fmb + r) * OLD + fn] = (bf16_t)c[m][r];
        }
    }
    __syncthreads();

    // ---- stage proj_w transposed bf16 (packed k-pairs; reuses R1) -----------
    for (int idx = tid; idx < CDIM * CDIM / 2; idx += THREADS) {
        int r2 = idx >> 7, c = idx & 127;
        int r  = 2 * r2;
        *(unsigned*)(pwT + c * PWTLD + r) =
            pack_bf16(proj_w[r * CDIM + c], proj_w[(r + 1) * CDIM + c]);
    }
    __syncthreads();

    // ---- GEMM4: proj + bias + exact GELU, store fp32 ------------------------
    {
        int nt = wave;                          // 0..7, B reused across 4 M-tiles
        v8f c[4];
#pragma unroll
        for (int m = 0; m < 4; ++m) c[m] = zero_v8f();
#pragma unroll
        for (int kp = 0; kp < 4; ++kp) {
            v16bf bf = load_frag_b_kmaj(pwT + nt * 16 * PWTLD + kp * 32, PWTLD);
#pragma unroll
            for (int m = 0; m < 4; ++m) {
                v16bf a = load_frag_a_row(obf + m * 16 * OLD + kp * 32, OLD);
                c[m] = wmma_bf16(a, bf, c[m]);
            }
        }
        int col   = nt * 16 + fn;
        float bia = proj_b[col];
        float* og = out + (size_t)b * NTOK * CDIM;
#pragma unroll
        for (int m = 0; m < 4; ++m) {
#pragma unroll
            for (int r = 0; r < 8; ++r) {
                int row = m * 16 + fmb + r;
                if (row < NTOK) {
                    float v = c[m][r] + bia;
                    og[row * CDIM + col] = 0.5f * v * (1.0f + erff(v * 0.7071067811865475f));
                }
            }
        }
    }
}

extern "C" void kernel_launch(void* const* d_in, const int* in_sizes, int n_in,
                              void* d_out, int out_size, void* d_ws, size_t ws_size,
                              hipStream_t stream) {
    (void)in_sizes; (void)n_in; (void)out_size; (void)d_ws; (void)ws_size;
    const float* x          = (const float*)d_in[0];
    const float* mask       = (const float*)d_in[1];
    const float* qkv_w      = (const float*)d_in[2];
    const float* qkv_b      = (const float*)d_in[3];
    const float* proj_w     = (const float*)d_in[4];
    const float* proj_b     = (const float*)d_in[5];
    const float* bias_table = (const float*)d_in[6];
    const int*   rel_idx    = (const int*)d_in[7];
    float* outp = (float*)d_out;

    // allow >64KB dynamic LDS (idempotent; not a stream op, capture-safe)
    hipFuncSetAttribute((const void*)swin_window_attn_kernel,
                        hipFuncAttributeMaxDynamicSharedMemorySize, LDS_TOTAL);

    swin_window_attn_kernel<<<dim3(4096), dim3(THREADS), LDS_TOTAL, stream>>>(
        x, mask, qkv_w, qkv_b, proj_w, proj_b, bias_table, rel_idx, outp);
}
